// GConvLSTM_46007689675315
// MI455X (gfx1250) — compile-verified, matrix operand order checked by reading, not run on hardware
//
#include <hip/hip_runtime.h>

// ---------------------------------------------------------------------------
// GConvLSTM on MI455X (gfx1250): bf16 WMMA 16x16x32 (f32 accum), all matrix
// operands pre-swizzled into WMMA fragment order so every fragment load is
// 32 contiguous bytes per lane (2 x b128), no per-element gathers.
//   B=8 T=64 N=512 F=64 H=128, gates = 4H = 512
// ---------------------------------------------------------------------------

typedef __attribute__((ext_vector_type(16))) __bf16 v16bf;
typedef __attribute__((ext_vector_type(2)))  __bf16 v2bf;
typedef __attribute__((ext_vector_type(8)))  float  v8f;

#define DEV __device__ __forceinline__

constexpr int BB = 8, TT = 64, NN = 512, FF = 64, HH = 128, GG = 512;
constexpr size_t HS_ELEMS = (size_t)BB * TT * NN * HH;  // hidden_seq elems
constexpr size_t ST_ELEMS = (size_t)BB * NN * HH;       // one (B,N,H) state
constexpr size_t HT_OFF   = HS_ELEMS;                   // h_T offset in d_out
constexpr size_t CT_OFF   = HS_ELEMS + ST_ELEMS;        // c_T offset in d_out

// frag-order block = 512 bf16 (1 KB): lane*16 + elem
DEV int lane_id() { return (int)(threadIdx.x & 31u); }
DEV int frag_lane(int mn, int kk) { return (mn & 15) + (((kk >> 3) & 1) << 4); }
DEV int frag_elem(int kk)         { return (kk & 7) + ((kk >> 4) << 3); }

DEV v16bf load_frag(const __bf16* block) {       // global or LDS, 2 x b128
  return ((const v16bf*)block)[lane_id()];
}

DEV v8f wmma_bf16(v16bf a, v16bf b, v8f c) {
  return __builtin_amdgcn_wmma_f32_16x16x32_bf16(
      false, a, false, b, (short)0, c, false, false);
}

DEV float sigmoid_f(float x) { return 1.0f / (1.0f + __expf(-x)); }

// ---------------------------------------------------------------------------
// pack: adj -> A-frag order bf16; U, W -> B-frag order bf16; zero h (x2), c
// adjF block (rt, ki) rt=rowtile<16>, ki=kchunk<32>:  (rt*16 + ki) * 512
// UF   block (ki, ct) ki=kchunk<32>,  ct=coltile<16>: (ki*32 + ct) * 512
// WF   block (ki, ct): (ki*32 + ct) * 512
// ---------------------------------------------------------------------------
__global__ void __launch_bounds__(256)
gclstm_pack(const float* __restrict__ adj,
            const float* __restrict__ U,
            const float* __restrict__ W,
            __bf16* __restrict__ adjF,
            __bf16* __restrict__ UF,
            __bf16* __restrict__ WF,
            __bf16* __restrict__ hbuf,
            float* __restrict__ c) {
  size_t i = (size_t)blockIdx.x * blockDim.x + threadIdx.x;
  if (i < (size_t)NN * NN) {               // adj (N x N), A-frag order
    int row = (int)(i >> 9), k = (int)(i & 511);
    int rt = row >> 4, m = row & 15, ki = k >> 5, kk = k & 31;
    adjF[(size_t)(rt * 16 + ki) * 512 + frag_lane(m, kk) * 16 + frag_elem(kk)] =
        (__bf16)adj[i];
  }
  if (i < (size_t)HH * GG) {               // U (H x 4H), B-frag order
    int K = (int)(i >> 9), col = (int)(i & 511);
    int ki = K >> 5, kk = K & 31, ct = col >> 4, n = col & 15;
    UF[(size_t)(ki * 32 + ct) * 512 + frag_lane(n, kk) * 16 + frag_elem(kk)] =
        (__bf16)U[i];
  }
  if (i < (size_t)FF * GG) {               // W (F x 4H), B-frag order
    int K = (int)(i >> 9), col = (int)(i & 511);
    int ki = K >> 5, kk = K & 31, ct = col >> 4, n = col & 15;
    WF[(size_t)(ki * 32 + ct) * 512 + frag_lane(n, kk) * 16 + frag_elem(kk)] =
        (__bf16)W[i];
  }
  if (i < 2 * ST_ELEMS) hbuf[i] = (__bf16)0.0f;
  if (i < ST_ELEMS) c[i] = 0.0f;
}

// ---------------------------------------------------------------------------
// axF[bt, rt, ki] = A-frag-order bf16 of (adj @ x[b,t]) rows [16rt,16rt+16)
// grid: (N/16, B*T), block 128 (4 waves; wave w = x coltile w)
// ---------------------------------------------------------------------------
__global__ void __launch_bounds__(128)
gclstm_ax(const __bf16* __restrict__ adjF,
          const float* __restrict__ x,
          __bf16* __restrict__ axF) {
  int bt = blockIdx.y;
  int rt = blockIdx.x;
  int tid = threadIdx.x;
  int wave = tid >> 5;
  const float* xbt = x + (size_t)bt * NN * FF;

  __shared__ __align__(32) __bf16 sXB[4 * 512];   // 32x64 x-tile, B-frag order
  v2bf* sXB2 = (v2bf*)sXB;

  v8f acc = {};
  for (int ki = 0; ki < 16; ++ki) {
    // stage x tile (32 x 64): row-pairs (kk, kk+1) of one column pack into
    // consecutive bf16 of one lane -> single b32 LDS store per pair.
    // 1024 pairs, exactly 8 per thread (no guards).
#pragma unroll
    for (int it = 0; it < 8; ++it) {
      int p = tid + it * 128;
      int r2 = p >> 6, cc = p & 63;
      int kk = r2 * 2;
      int ct = cc >> 4, n = cc & 15;
      float lo = xbt[(size_t)(ki * 32 + kk) * FF + cc];
      float hi = xbt[(size_t)(ki * 32 + kk + 1) * FF + cc];
      v2bf pr;
      pr[0] = (__bf16)lo;
      pr[1] = (__bf16)hi;
      sXB2[(ct * 512 + frag_lane(n, kk) * 16 + frag_elem(kk)) >> 1] = pr;
    }
    __syncthreads();
    v16bf a = load_frag(adjF + (size_t)(rt * 16 + ki) * 512);
    v16bf b = load_frag(sXB + wave * 512);
    acc = wmma_bf16(a, b, acc);
    __syncthreads();
  }
  // store C -> axF in A-frag order (K dim = F = 64 -> 2 chunks)
  int l = lane_id(), n = l & 15, half = l >> 4;
#pragma unroll
  for (int r = 0; r < 8; ++r) {
    int m = r + 8 * half;
    int col = wave * 16 + n;
    int ki = col >> 5, kk = col & 31;
    axF[((size_t)(bt * 32 + rt) * 2 + ki) * 512 + frag_lane(m, kk) * 16 +
        frag_elem(kk)] = (__bf16)acc[r];
  }
}

// ---------------------------------------------------------------------------
// One recurrence step. grid: (N/16, B), block 256 (8 waves).
//   phase1: tmp = adj_rows @ h_b   (A,B frags straight from global, no LDS)
//   phase2: gates = tmp @ U + ax @ W  (tmp via 4 KB LDS in A-frag order)
//   phase3: fused LSTM two rows at a time; h written packed in B-frag order
// hF block (b, ki<32>, ct<16>): b*64K + (ki*8 + ct)*512
// ---------------------------------------------------------------------------
__global__ void __launch_bounds__(256)
gclstm_step(int t,
            const __bf16* __restrict__ adjF,
            const __bf16* __restrict__ UF,
            const __bf16* __restrict__ WF,
            const float* __restrict__ bias,
            const __bf16* __restrict__ axF,
            const __bf16* __restrict__ hF_in,
            __bf16* __restrict__ hF_out,
            float* __restrict__ c_ws,
            float* __restrict__ out) {
  int b = blockIdx.y;
  int rt = blockIdx.x;
  int row0 = rt * 16;
  int tid = threadIdx.x;
  int wave = tid >> 5;
  int l = lane_id(), n = l & 15, half = l >> 4;
  int bt = b * TT + t;

  __shared__ __align__(32) __bf16 sTmp[4 * 512];    // 16x128 tmp, A-frag order
  __shared__ __align__(16) float  sGates[16 * GG];  // 32 KB

  // warm L2 -> WGP path for phase-2 operands while phase 1 runs
  __builtin_prefetch(axF + ((size_t)(bt * 32 + rt) * 2) * 512, 0, 3);
  __builtin_prefetch(UF + (size_t)(wave * 4) * 512, 0, 3);

  // ---- phase 1: tmp = adj[rows] @ h_b ; wave w owns tmp coltile w ----
  const __bf16* hb = hF_in + (size_t)b * NN * HH;
  v8f acc0 = {}, acc1 = {};
#pragma unroll
  for (int ki = 0; ki < 16; ki += 2) {
    v16bf a0 = load_frag(adjF + (size_t)(rt * 16 + ki) * 512);
    v16bf b0 = load_frag(hb + (size_t)(ki * 8 + wave) * 512);
    v16bf a1 = load_frag(adjF + (size_t)(rt * 16 + ki + 1) * 512);
    v16bf b1 = load_frag(hb + (size_t)((ki + 1) * 8 + wave) * 512);
    acc0 = wmma_bf16(a0, b0, acc0);
    acc1 = wmma_bf16(a1, b1, acc1);
  }
  // C(m = r+8*half, col = wave*16+n) -> sTmp A-frag order (K = 128)
#pragma unroll
  for (int r = 0; r < 8; ++r) {
    int m = r + 8 * half;
    int col = wave * 16 + n;
    int ki = col >> 5, kk = col & 31;
    sTmp[ki * 512 + frag_lane(m, kk) * 16 + frag_elem(kk)] =
        (__bf16)(acc0[r] + acc1[r]);
  }
  __syncthreads();

  // ---- phase 2: gates = tmp @ U + ax @ W ; wave w owns coltiles 4w..4w+3 --
  v8f gc[4] = {{}, {}, {}, {}};
#pragma unroll
  for (int ki = 0; ki < 4; ++ki) {
    v16bf a = load_frag(sTmp + ki * 512);
#pragma unroll
    for (int j = 0; j < 4; ++j) {
      int ct = wave * 4 + j;
      v16bf bb = load_frag(UF + (size_t)(ki * 32 + ct) * 512);
      gc[j] = wmma_bf16(a, bb, gc[j]);
    }
  }
#pragma unroll
  for (int ki = 0; ki < 2; ++ki) {
    v16bf a = load_frag(axF + ((size_t)(bt * 32 + rt) * 2 + ki) * 512);
#pragma unroll
    for (int j = 0; j < 4; ++j) {
      int ct = wave * 4 + j;
      v16bf bb = load_frag(WF + (size_t)(ki * 32 + ct) * 512);
      gc[j] = wmma_bf16(a, bb, gc[j]);
    }
  }
  __syncthreads();
#pragma unroll
  for (int j = 0; j < 4; ++j) {
    int n0 = (wave * 4 + j) * 16;
#pragma unroll
    for (int r = 0; r < 8; ++r)
      sGates[(r + 8 * half) * GG + n0 + n] = gc[j][r];
  }
  __syncthreads();

  // ---- phase 3: fused LSTM, two rows per item (1024 items, 4/thread) ----
  float* cb = c_ws + (size_t)b * NN * HH;
  __bf16* ho = hF_out + (size_t)b * NN * HH;
  v2bf* ho2 = (v2bf*)ho;
#pragma unroll
  for (int it = 0; it < 4; ++it) {
    int e = tid + it * 256;
    int r2 = e >> 7, hc = e & 127;
    int r = r2 * 2;
    float bi = bias[hc], bf = bias[128 + hc], bg = bias[256 + hc],
          bo = bias[384 + hc];

    float h01[2], c01[2];
#pragma unroll
    for (int q = 0; q < 2; ++q) {
      int rr = r + q;
      float iv = sigmoid_f(sGates[rr * GG + hc] + bi);
      float fv = sigmoid_f(sGates[rr * GG + 128 + hc] + bf);
      float gv = tanhf(sGates[rr * GG + 256 + hc] + bg);
      float ov = sigmoid_f(sGates[rr * GG + 384 + hc] + bo);
      size_t ci = (size_t)(row0 + rr) * HH + hc;
      float cnew = fv * cb[ci] + iv * gv;
      float hnew = ov * tanhf(cnew);
      cb[ci] = cnew;
      c01[q] = cnew;
      h01[q] = hnew;
      out[((size_t)bt * NN + row0 + rr) * HH + hc] = hnew;
    }
    // packed h store in B-frag order: rows (r, r+1) -> one lane, 2 elems
    int nrow = row0 + r;
    int ki = nrow >> 5, kk = nrow & 31, ct = hc >> 4, nn = hc & 15;
    v2bf pr;
    pr[0] = (__bf16)h01[0];
    pr[1] = (__bf16)h01[1];
    ho2[((size_t)(ki * 8 + ct) * 512 + frag_lane(nn, kk) * 16 + frag_elem(kk)) >> 1] = pr;
    if (t == TT - 1) {
#pragma unroll
      for (int q = 0; q < 2; ++q) {
        size_t ci = (size_t)(row0 + r + q) * HH + hc;
        out[HT_OFF + (size_t)b * NN * HH + ci] = h01[q];
        out[CT_OFF + (size_t)b * NN * HH + ci] = c01[q];
      }
    }
  }
}

// ---------------------------------------------------------------------------
extern "C" void kernel_launch(void* const* d_in, const int* in_sizes, int n_in,
                              void* d_out, int out_size, void* d_ws, size_t ws_size,
                              hipStream_t stream) {
  const float* x    = (const float*)d_in[0];  // (B,T,N,F)
  const float* adj  = (const float*)d_in[1];  // (N,N)
  const float* Wm   = (const float*)d_in[2];  // (F,4H)
  const float* Um   = (const float*)d_in[3];  // (H,4H)
  const float* bias = (const float*)d_in[4];  // (4H)
  float* out = (float*)d_out;

  // workspace layout (bytes), all frag-order bf16 unless noted
  char* p = (char*)d_ws;
  __bf16* adjF = (__bf16*)p;                 p += (size_t)NN * NN * 2;     // 512K
  __bf16* UF   = (__bf16*)p;                 p += (size_t)HH * GG * 2;     // 128K
  __bf16* WF   = (__bf16*)p;                 p += (size_t)FF * GG * 2;     //  64K
  __bf16* axF  = (__bf16*)p;                 p += (size_t)BB*TT*NN*FF * 2; // 32M
  __bf16* hbuf = (__bf16*)p;                 p += 2 * ST_ELEMS * 2;        //  2M
  float*  c_ws = (float*)p;                                               //  2M (f32)

  // 1) pack weights/adj to frag order, zero recurrent state
  {
    size_t nmax = 2 * ST_ELEMS;  // 1,048,576 >= all pack ranges
    gclstm_pack<<<dim3((unsigned)((nmax + 255) / 256)), 256, 0, stream>>>(
        adj, Um, Wm, adjF, UF, WF, hbuf, c_ws);
  }
  // 2) hoisted x-path: axF = frag(adj @ x_t), parallel over all B*T slices
  gclstm_ax<<<dim3(NN / 16, BB * TT), 128, 0, stream>>>(adjF, x, axF);

  // 3) sequential recurrence, stream-ordered (global barrier per step)
  for (int t = 0; t < TT; ++t) {
    __bf16* h_in  = hbuf + (size_t)(t & 1) * ST_ELEMS;
    __bf16* h_out = hbuf + (size_t)((t + 1) & 1) * ST_ELEMS;
    gclstm_step<<<dim3(NN / 16, BB), 256, 0, stream>>>(
        t, adjF, UF, WF, bias, axF, h_in, h_out, c_ws, out);
  }
}
